// mySAGE_88003879895212
// MI455X (gfx1250) — matrix operand order repaired and checked by reading, not compiled, not run on hardware
//
#include <hip/hip_runtime.h>
#include <hip/hip_bf16.h>

// ---------------------------------------------------------------------------
// ALIGNN-style crystal GNN for MI455X (gfx1250, wave32, WMMA bf16)
//   - GBF expansion fused into WMMA A-tile staging (LDS, fragment layout)
//   - Weights pre-packed to bf16 fragment order (contiguous 32B/lane loads)
//   - Each wave computes a 16x32 output tile: one A fragment feeds two WMMAs
//   - scatter-mean via global f32 atomics (L2-resident working set)
// ---------------------------------------------------------------------------

typedef __attribute__((ext_vector_type(16))) __bf16 v16bf;
typedef __attribute__((ext_vector_type(8)))  float  v8f;

#define NEIGH 12
#define HID   128

// A-fragment K offset for 16-bit WMMA (16x32), per ISA 7.12.2:
// lanes 0-15 hold K {0..7,16..23}, lanes 16-31 hold K {8..15,24..31}
__device__ __forceinline__ int koffA(int e, int half) {
  int j = e >> 1, w = e & 1;
  return (j < 4) ? (8 * half + 2 * j + w) : (16 + 8 * half + 2 * (j - 4) + w);
}

// ---------------------------------------------------------------------------
// Weight packing: fp32 [128,K] (lin_l) and [128,K] (lin_r) -> bf16 fragments
// packed[ct][kk][lane][e], ct in 0..15 (256 output cols), B-layout:
// lane = n%16 + 16*(K half), K order matches A fragment (koffA).
// ---------------------------------------------------------------------------
template <int KSTEPS>
__global__ void pack_weights(const float* __restrict__ Wl,
                             const float* __restrict__ Wr,
                             __bf16* __restrict__ packed) {
  const int KTOT = KSTEPS * 32;
  const int total = 16 * KSTEPS * 512;
  int idx = blockIdx.x * 256 + threadIdx.x;
  if (idx >= total) return;
  int ct   = idx / (KSTEPS * 512);
  int rem  = idx - ct * (KSTEPS * 512);
  int kk   = rem >> 9;
  int lane = (rem >> 4) & 31;
  int e    = rem & 15;
  int n    = lane & 15;
  int half = lane >> 4;
  int gcol = ct * 16 + n;
  int K    = kk * 32 + koffA(e, half);
  float w = (gcol < HID) ? Wl[(size_t)gcol * KTOT + K]
                         : Wr[(size_t)(gcol - HID) * KTOT + K];
  packed[idx] = (__bf16)w;
}

// ---------------------------------------------------------------------------
// Fused (GBF ->) bf16 WMMA GEMM: out[16 x 256] per row-tile, one block each.
// 8 waves x 2 col-tiles per wave = 16 col-tiles (256 combined output cols).
// MODE 0: bond GBF  (src = f32 [N,12],  K = 12*40  = 480,  KSTEPS = 15)
// MODE 1: angle GBF (src = f32 [N,144], K = 144*10 = 1440, KSTEPS = 45)
// MODE 2: plain bf16 activations (src = bf16 [N,128], KSTEPS = 4)
// Cols 0..127 -> hl (lin_l pre-aggregation), 128..255 -> hr (lin_r term).
// ---------------------------------------------------------------------------
template <int KSTEPS, int MODE>
__global__ __launch_bounds__(256) void gemm_sage(
    const void* __restrict__ srcv, const __bf16* __restrict__ packedW,
    float* __restrict__ hl, float* __restrict__ hr, int natoms) {
  extern __shared__ __align__(32) __bf16 ldsA[];  // KSTEPS*512 bf16
  const int tid  = threadIdx.x;
  const int base = blockIdx.x * 16;

  // Stage A tile into LDS directly in WMMA fragment layout (with fused GBF).
  for (int t = tid; t < KSTEPS * 512; t += 256) {
    int kk   = t >> 9;
    int lane = (t >> 4) & 31;
    int e    = t & 15;
    int row  = lane & 15;
    int half = lane >> 4;
    int K    = kk * 32 + koffA(e, half);
    int atom = base + row;
    if (atom >= natoms) atom = natoms - 1;
    float val;
    if (MODE == 0) {
      const float* bf = (const float*)srcv;
      int j = K / 40, s = K - j * 40;
      float d  = bf[(size_t)atom * NEIGH + j];
      float mu = 8.0f * (float)s / 39.0f;       // linspace(0,8,40)
      float df = d - mu;
      val = __expf(-25.0f * df * df);           // gamma = 0.2
    } else if (MODE == 1) {
      const float* af = (const float*)srcv;
      int p = K / 10, s = K - p * 10;
      float d  = af[(size_t)atom * (NEIGH * NEIGH) + p];
      float mu = -1.0f + (2.0f / 9.0f) * (float)s;  // linspace(-1,1,10)
      float df = d - mu;
      val = __expf(-25.0f * df * df);           // gamma = 0.2
    } else {
      const __bf16* xf = (const __bf16*)srcv;
      val = (float)xf[(size_t)atom * HID + K];
    }
    ldsA[t] = (__bf16)val;
  }
  __syncthreads();

  const int lane = tid & 31;
  const int wave = tid >> 5;
  const int ct0  = wave * 2;  // two adjacent col-tiles per wave

  v8f acc0 = {0.f, 0.f, 0.f, 0.f, 0.f, 0.f, 0.f, 0.f};
  v8f acc1 = {0.f, 0.f, 0.f, 0.f, 0.f, 0.f, 0.f, 0.f};
  const v16bf* wp0 = (const v16bf*)packedW + (size_t)ct0 * KSTEPS * 32;
  const v16bf* wp1 = wp0 + (size_t)KSTEPS * 32;
  for (int kk = 0; kk < KSTEPS; ++kk) {
    v16bf a  = *(const v16bf*)(&ldsA[kk * 512 + lane * 16]);  // ds_load_b128 x2
    v16bf b0 = wp0[kk * 32 + lane];                           // global 32B/lane
    v16bf b1 = wp1[kk * 32 + lane];
    acc0 = __builtin_amdgcn_wmma_f32_16x16x32_bf16(
        false, a, false, b0, (short)0, acc0, false, false);
    acc1 = __builtin_amdgcn_wmma_f32_16x16x32_bf16(
        false, a, false, b1, (short)0, acc1, false, false);
  }

  // C/D layout: n = lane%16, m = vgpr + 8*(lane>=16)
  const int n  = lane & 15;
  const int hc = lane >> 4;
#pragma unroll
  for (int half_ct = 0; half_ct < 2; ++half_ct) {
    const int gcol = (ct0 + half_ct) * 16 + n;
    float* dst = (gcol < HID) ? (hl + gcol) : (hr + (gcol - HID));
    const v8f& acc = half_ct ? acc1 : acc0;
#pragma unroll
    for (int v = 0; v < 8; ++v) {
      int atom = base + v + 8 * hc;
      if (atom < natoms) dst[(size_t)atom * HID] = acc[v];
    }
  }
}

// ---------------------------------------------------------------------------
__global__ void fill_zero(float* __restrict__ p, long n) {
  long i = (long)blockIdx.x * 256 + threadIdx.x;
  if (i < n) p[i] = 0.f;
}

__global__ void count_deg(const int* __restrict__ nbr, float* __restrict__ cnt,
                          int nedges) {
  int e = blockIdx.x * 256 + threadIdx.x;
  if (e < nedges) atomicAdd(&cnt[nbr[e]], 1.0f);
}

// s[d] += h[i] for each neighbor d of atom i (2 atoms per 256-thread block)
__global__ void scatter_add(const float* __restrict__ h,
                            const int* __restrict__ nbr,
                            float* __restrict__ s, int natoms) {
  int atom = blockIdx.x * 2 + (threadIdx.x >> 7);
  int c = threadIdx.x & 127;
  if (atom >= natoms) return;
  float v = h[(size_t)atom * HID + c];
#pragma unroll
  for (int j = 0; j < NEIGH; ++j) {
    int d = nbr[atom * NEIGH + j];
    atomicAdd(&s[(size_t)d * HID + c], v);
  }
}

// act = relu(s/max(cnt,1) + bias + r); emit bf16 (next GEMM) and/or f32 (pool)
template <bool WB, bool WF>
__global__ void combine_act(const float* __restrict__ s,
                            const float* __restrict__ r,
                            const float* __restrict__ cnt,
                            const float* __restrict__ bias,
                            __bf16* __restrict__ obf, float* __restrict__ of,
                            int natoms) {
  int atom = blockIdx.x * 2 + (threadIdx.x >> 7);
  int c = threadIdx.x & 127;
  if (atom >= natoms) return;
  float cc = cnt[atom];
  cc = cc < 1.f ? 1.f : cc;
  size_t i = (size_t)atom * HID + c;
  float v = s[i] / cc + bias[c] + r[i];
  v = v > 0.f ? v : 0.f;
  if (WB) obf[i] = (__bf16)v;
  if (WF) of[i] = v;
}

// segment-mean over contiguous crystal range + 256->2 FC
__global__ void pool_fc(const float* __restrict__ bo, const float* __restrict__ ao,
                        const int* __restrict__ crys, const float* __restrict__ Wfc,
                        const float* __restrict__ bfc, float* __restrict__ out) {
  __shared__ float p[2 * HID];
  int b = blockIdx.x;
  int c = threadIdx.x;  // 0..255
  int st = crys[b * 2 + 0];
  int en = crys[b * 2 + 1];
  const float* src = (c < HID) ? bo : ao;
  int cc = c & (HID - 1);
  float acc = 0.f;
  for (int a = st; a < en; ++a) acc += src[(size_t)a * HID + cc];
  float n = (float)(en - st);
  p[c] = acc / (n < 1.f ? 1.f : n);
  __syncthreads();
  if (c < 2) {
    float o = bfc[c];
    for (int k = 0; k < 2 * HID; ++k) o += p[k] * Wfc[c * (2 * HID) + k];
    out[b * 2 + c] = o;
  }
}

// ---------------------------------------------------------------------------
extern "C" void kernel_launch(void* const* d_in, const int* in_sizes, int n_in,
                              void* d_out, int out_size, void* d_ws,
                              size_t ws_size, hipStream_t stream) {
  const float* bond  = (const float*)d_in[0];
  const float* angle = (const float*)d_in[1];
  // d_in[2] = species (unused by the reference graph math)
  const int*   nbr   = (const int*)d_in[3];
  const int*   crys  = (const int*)d_in[4];
  const float* W1b   = (const float*)d_in[5];
  const float* b1b   = (const float*)d_in[6];
  const float* W1br  = (const float*)d_in[7];
  const float* W1a   = (const float*)d_in[8];
  const float* b1a   = (const float*)d_in[9];
  const float* W1ar  = (const float*)d_in[10];
  const float* W2b   = (const float*)d_in[11];
  const float* b2b   = (const float*)d_in[12];
  const float* W2br  = (const float*)d_in[13];
  const float* W2a   = (const float*)d_in[14];
  const float* b2a   = (const float*)d_in[15];
  const float* W2ar  = (const float*)d_in[16];
  const float* Wfc   = (const float*)d_in[17];
  const float* bfc   = (const float*)d_in[18];
  float* out = (float*)d_out;

  const int natoms = in_sizes[0] / NEIGH;
  const int B      = in_sizes[4] / 2;
  const int nedges = natoms * NEIGH;

  char* ws = (char*)d_ws;
  size_t off = 0;
  auto alloc = [&](size_t bytes) -> void* {
    void* p = ws + off;
    off = (off + bytes + 255) & ~(size_t)255;
    return p;
  };

  __bf16* pW1b = (__bf16*)alloc((size_t)16 * 15 * 512 * 2);
  __bf16* pW1a = (__bf16*)alloc((size_t)16 * 45 * 512 * 2);
  __bf16* pW2b = (__bf16*)alloc((size_t)16 * 4 * 512 * 2);
  __bf16* pW2a = (__bf16*)alloc((size_t)16 * 4 * 512 * 2);
  float*  cnt  = (float*)alloc((size_t)natoms * 4);
  float*  hl   = (float*)alloc((size_t)natoms * HID * 4);
  float*  hr   = (float*)alloc((size_t)natoms * HID * 4);
  float*  sbuf = (float*)alloc((size_t)natoms * HID * 4);
  __bf16* actb = (__bf16*)alloc((size_t)natoms * HID * 2);
  __bf16* acta = (__bf16*)alloc((size_t)natoms * HID * 2);
  float*  bo2  = (float*)alloc((size_t)natoms * HID * 4);
  float*  ao2  = (float*)alloc((size_t)natoms * HID * 4);

  const int grows = (natoms + 15) / 16;
  const int nb2 = (natoms + 1) / 2;
  const long nf = (long)natoms * HID;
  const int zf = (int)((nf + 255) / 256);
  const int zc = (natoms + 255) / 256;

  // ---- pack weights (bf16, fragment order) --------------------------------
  pack_weights<15><<<(16 * 15 * 512 + 255) / 256, 256, 0, stream>>>(W1b, W1br, pW1b);
  pack_weights<45><<<(16 * 45 * 512 + 255) / 256, 256, 0, stream>>>(W1a, W1ar, pW1a);
  pack_weights<4><<<(16 * 4 * 512 + 255) / 256, 256, 0, stream>>>(W2b, W2br, pW2b);
  pack_weights<4><<<(16 * 4 * 512 + 255) / 256, 256, 0, stream>>>(W2a, W2ar, pW2a);

  // ---- in-degree counts (graph-invariant across all 4 convs) --------------
  fill_zero<<<zc, 256, 0, stream>>>(cnt, natoms);
  count_deg<<<(nedges + 255) / 256, 256, 0, stream>>>(nbr, cnt, nedges);

  // ---- bond conv 1: GBF(480) fused GEMM -> scatter-mean -> relu -----------
  gemm_sage<15, 0><<<grows, 256, 15 * 512 * 2, stream>>>(bond, pW1b, hl, hr, natoms);
  fill_zero<<<zf, 256, 0, stream>>>(sbuf, nf);
  scatter_add<<<nb2, 256, 0, stream>>>(hl, nbr, sbuf, natoms);
  combine_act<true, false><<<nb2, 256, 0, stream>>>(sbuf, hr, cnt, b1b, actb, nullptr, natoms);

  // ---- angle conv 1: GBF(1440) fused GEMM ---------------------------------
  gemm_sage<45, 1><<<grows, 256, 45 * 512 * 2, stream>>>(angle, pW1a, hl, hr, natoms);
  fill_zero<<<zf, 256, 0, stream>>>(sbuf, nf);
  scatter_add<<<nb2, 256, 0, stream>>>(hl, nbr, sbuf, natoms);
  combine_act<true, false><<<nb2, 256, 0, stream>>>(sbuf, hr, cnt, b1a, acta, nullptr, natoms);

  // ---- bond conv 2: bf16 128-K GEMM ---------------------------------------
  gemm_sage<4, 2><<<grows, 256, 4 * 512 * 2, stream>>>(actb, pW2b, hl, hr, natoms);
  fill_zero<<<zf, 256, 0, stream>>>(sbuf, nf);
  scatter_add<<<nb2, 256, 0, stream>>>(hl, nbr, sbuf, natoms);
  combine_act<false, true><<<nb2, 256, 0, stream>>>(sbuf, hr, cnt, b2b, nullptr, bo2, natoms);

  // ---- angle conv 2 -------------------------------------------------------
  gemm_sage<4, 2><<<grows, 256, 4 * 512 * 2, stream>>>(acta, pW2a, hl, hr, natoms);
  fill_zero<<<zf, 256, 0, stream>>>(sbuf, nf);
  scatter_add<<<nb2, 256, 0, stream>>>(hl, nbr, sbuf, natoms);
  combine_act<false, true><<<nb2, 256, 0, stream>>>(sbuf, hr, cnt, b2a, nullptr, ao2, natoms);

  // ---- segment-mean pooling + FC ------------------------------------------
  pool_fc<<<B, 256, 0, stream>>>(bo2, ao2, crys, Wfc, bfc, out);
}
